// GRUD_3461743641008
// MI455X (gfx1250) — matrix operand order repaired and checked by reading, not compiled, not run on hardware
//
#include <hip/hip_runtime.h>
#include <hip/hip_bf16.h>
#include <math.h>
#include <stdint.h>

typedef __attribute__((ext_vector_type(16))) __bf16 v16bf;
typedef __attribute__((ext_vector_type(8)))  __bf16 v8bf;
typedef __attribute__((ext_vector_type(8)))  float  v8f;
typedef __attribute__((ext_vector_type(4))) unsigned int u32x4;
typedef __attribute__((ext_vector_type(8))) int          i32x8;
typedef __attribute__((ext_vector_type(4))) int          i32x4;

// global (address_space(1)) views of the bf16 weights: keeps loads as global_load_b128
typedef __attribute__((address_space(1))) const __bf16 gbf;
typedef __attribute__((address_space(1))) const v16bf gv16bf;

constexpr int kB  = 256;   // batch
constexpr int kT  = 512;   // time steps
constexpr int kD  = 256;   // feature dim
constexpr int kH  = 256;   // hidden dim
constexpr int kK  = 768;   // D + H + D (comb width)
constexpr int kMT = 16;    // batch rows per workgroup (WMMA M)

constexpr int kAPAD = kK + 8;   // 776 (pad keeps 16B align: 776*2 = 97*16)
constexpr int kDPAD = kD + 8;   // 264

// bf16 weight layout inside d_ws (element offsets)
constexpr int kWghOff = 0;
constexpr int kWghN   = kD * kD;          // 65536
constexpr int kWzOff  = kWghOff + kWghN;
constexpr int kWRow   = kH * kK;          // 196608
constexpr int kWrOff  = kWzOff + kWRow;
constexpr int kWhOff  = kWrOff + kWRow;
constexpr int kWTot   = kWhOff + kWRow;   // 655360 elems = 1.25 MB bf16

constexpr unsigned kDynLDS = 2u * 4u * kMT * kD * sizeof(float);  // 128 KB input double buffer

// ---------------- prep: fp32 -> bf16 weight conversion into workspace --------
__global__ void grud_prep(const float* __restrict__ Wg_h, const float* __restrict__ Wz,
                          const float* __restrict__ Wr,   const float* __restrict__ Wh,
                          __bf16* __restrict__ wsb) {
    for (int i = blockIdx.x * blockDim.x + threadIdx.x; i < kWTot;
         i += gridDim.x * blockDim.x) {
        float v;
        if      (i < kWzOff) v = Wg_h[i - kWghOff];
        else if (i < kWrOff) v = Wz[i - kWzOff];
        else if (i < kWhOff) v = Wr[i - kWrOff];
        else                 v = Wh[i - kWhOff];
        wsb[i] = (__bf16)v;
    }
}

// A fragment (16x32 bf16): lane l -> row l%16, 16 contiguous K at (l/16)*16 (matches B)
__device__ __forceinline__ v16bf load_a_lds(const __bf16* row0, int stride, int kbase, int lane) {
    const __bf16* p = row0 + (lane & 15) * stride + kbase + ((lane >> 4) << 4);
    v8bf lo = *(const v8bf*)(p);
    v8bf hi = *(const v8bf*)(p + 8);
    return __builtin_shufflevector(lo, hi, 0,1,2,3,4,5,6,7,8,9,10,11,12,13,14,15);
}

// B fragment (32x16 bf16): lane l -> weight row (c0 + l%16), 16 contiguous K at kbase + (l/16)*16
__device__ __forceinline__ v16bf load_b_glb(gbf* W, int ldk, int c0, int kbase, int lane) {
    gbf* p = W + (size_t)(c0 + (lane & 15)) * ldk + kbase + ((lane >> 4) << 4);
    return *(gv16bf*)p;   // 32B aligned -> 2x global_load_b128 from L2-resident weights
}

#define WMMA_BF16(A, Bm, C) \
    __builtin_amdgcn_wmma_f32_16x16x32_bf16(false, (A), false, (Bm), (short)0, (C), false, false)

// TDM: async-load one 16x256 f32 tile (row stride 4*T*D elems) from global into LDS.
// 6-arg builtin on this toolchain: (u32x4 g0, i32x8 g1, i32x4 g2, i32x4 g3, i32x8 g4, i32 cpol)
__device__ __forceinline__ void tdm_load_tile(const float* gsrc, void* ldst) {
    uint64_t ga = (uint64_t)(uintptr_t)gsrc;
    u32x4 g0;
    g0[0] = 1u;                                            // count=1, user mode
    g0[1] = (unsigned)(uintptr_t)ldst;                     // lds_addr (low 32 bits of flat = LDS offset)
    g0[2] = (unsigned)(ga & 0xffffffffu);                  // global_addr[31:0]
    g0[3] = (unsigned)((ga >> 32) & 0x01ffffffu) | (2u << 30);  // global_addr[56:32] | type=2
    i32x8 g1;
    g1[0] = (int)(0x2u << 16);                             // data_size = 4 bytes
    g1[1] = (int)(256u << 16);                             // tensor_dim0[15:0]=256  (abar_addr=0)
    g1[2] = (int)(16u  << 16);                             // tensor_dim0 hi=0 | tensor_dim1 lo=16
    g1[3] = (int)(256u << 16);                             // tensor_dim1 hi=0 | tile_dim0=256
    g1[4] = 16;                                            // tile_dim1=16, tile_dim2=0
    g1[5] = (int)(4u * kT * kD);                           // tensor_dim0_stride = 524288 elems
    g1[6] = 0;                                             // stride hi | dim1_stride lo
    g1[7] = 0;
    i32x4 z4 = {0, 0, 0, 0};
    i32x8 z8 = {0, 0, 0, 0, 0, 0, 0, 0};
    __builtin_amdgcn_tensor_load_to_lds(g0, g1, z4, z4, z8, 0);
}

// ---------------- persistent GRU-D kernel: one WG = 16 batch rows, all 512 steps ----------
__global__ __launch_bounds__(256) void grud_main(
    const float* __restrict__ inp,   const float* __restrict__ X_mean,
    const float* __restrict__ Wg_x,  const float* __restrict__ bg_x,
    const float* __restrict__ bg_h,  const float* __restrict__ bz,
    const float* __restrict__ br,    const float* __restrict__ bh,
    const float* __restrict__ Wfc,   const float* __restrict__ bfc,
    const __bf16* __restrict__ wsb,  float* __restrict__ out)
{
    __shared__ __bf16 sA[kMT][kAPAD];   // comb = [x | h | m]  (bf16)
    __shared__ __bf16 sM2[kMT][kDPAD];  // r*h middle slice    (bf16)
    __shared__ __bf16 sD[kMT][kDPAD];   // delta d             (bf16)
    __shared__ float  sH[kMT][kH];      // recurrent state     (fp32)
    __shared__ float  sDiag[kD], sBgx[kD], sXmean[kD];
    extern __shared__ float sIn[];      // [2][4][kMT][kD] input double buffer (TDM target)

    const int tid  = threadIdx.x;
    const int lane = tid & 31;
    const int w    = tid >> 5;          // wave 0..7, owns H cols [w*32, w*32+32)
    const int c0   = w * 32;            // first of the wave's two 16-col tiles
    const int b0   = blockIdx.x * kMT;

    // one-time constants
    sDiag[tid]  = Wg_x[tid * kD + tid];
    sBgx[tid]   = bg_x[tid];
    sXmean[tid] = X_mean[tid];
    for (int i = tid; i < kMT * kH; i += blockDim.x) sH[i >> 8][i & 255] = 0.f;

    float c_bgh[2], c_bz[2], c_br[2], c_bh[2];
#pragma unroll
    for (int tt = 0; tt < 2; ++tt) {
        const int col = c0 + tt * 16 + (lane & 15);
        c_bgh[tt] = bg_h[col]; c_bz[tt] = bz[col]; c_br[tt] = br[col]; c_bh[tt] = bh[col];
    }

    // kick off TDM for ts=0: waves 0..3 each DMA one input channel tile into buffer 0
    if (w < 4) {
        const float* gsrc = inp + ((size_t)b0 * 4 + w) * kT * kD;   // (b0, ch=w, ts=0, 0)
        tdm_load_tile(gsrc, &sIn[(size_t)w * kMT * kD]);
    }
    __syncthreads();

    float hn[2][8];   // delta_h * h (this step), regs per wave tile
    float zz[2][8];   // update gate

    for (int ts = 0; ts < kT; ++ts) {
        // re-materialize weight bases each step (LICM break) but cast back to
        // address_space(1) so loads stay global_load_b128 with cheap addressing.
        uintptr_t pgh = (uintptr_t)(wsb + kWghOff);
        uintptr_t pz  = (uintptr_t)(wsb + kWzOff);
        uintptr_t pr  = (uintptr_t)(wsb + kWrOff);
        uintptr_t ph  = (uintptr_t)(wsb + kWhOff);
        asm volatile("" : "+s"(pgh), "+s"(pz), "+s"(pr), "+s"(ph));
        gbf* WghB = (gbf*)pgh;
        gbf* WzB  = (gbf*)pz;
        gbf* WrB  = (gbf*)pr;
        gbf* WhB  = (gbf*)ph;

        // ---- phase 0: wait for this step's TDM tiles, elementwise front-end ----
        __builtin_amdgcn_s_wait_tensorcnt(0);   // own wave's DMA done (no-op for waves 4..7)
        __syncthreads();                         // all channels visible to all waves
        {
            const float* bufp = sIn + (size_t)(ts & 1) * 4 * kMT * kD;
            const int col = tid;                 // 256 threads == D columns
#pragma unroll 4
            for (int r = 0; r < kMT; ++r) {
                const float x  = bufp[(0 * kMT + r) * kD + col];
                const float xl = bufp[(1 * kMT + r) * kD + col];
                const float m  = bufp[(2 * kMT + r) * kD + col];
                const float d  = bufp[(3 * kMT + r) * kD + col];
                const float dx = __expf(-fmaxf(d * sDiag[col] + sBgx[col], 0.f));
                const float xn = m * x + (1.f - m) * (dx * xl + (1.f - dx) * sXmean[col]);
                sA[r][col]          = (__bf16)xn;
                sA[r][2 * kD + col] = (__bf16)m;
                sD[r][col]          = (__bf16)d;
            }
        }
        // prefetch next step's inputs into the other buffer while GEMMs run
        if (w < 4 && (ts + 1) < kT) {
            const float* gsrc = inp + ((size_t)b0 * 4 + w) * kT * kD + (size_t)(ts + 1) * kD;
            tdm_load_tile(gsrc, &sIn[((size_t)((ts + 1) & 1) * 4 + w) * kMT * kD]);
        }
        __syncthreads();

        // ---- phase 1: delta_h = exp(-relu(d @ Wg_h.T + bg_h)); h *= delta_h ----
        {
            v8f acc[2] = {};
#pragma unroll
            for (int kc = 0; kc < 8; ++kc) {
                v16bf a = load_a_lds(&sD[0][0], kDPAD, kc * 32, lane);   // shared by both tiles
                acc[0] = WMMA_BF16(a, load_b_glb(WghB, kD, c0,      kc * 32, lane), acc[0]);
                acc[1] = WMMA_BF16(a, load_b_glb(WghB, kD, c0 + 16, kc * 32, lane), acc[1]);
            }
#pragma unroll
            for (int tt = 0; tt < 2; ++tt) {
                const int col = c0 + tt * 16 + (lane & 15);
                const int rb  = (lane >> 4) << 3;
#pragma unroll
                for (int v = 0; v < 8; ++v) {
                    const float dh = __expf(-fmaxf(acc[tt][v] + c_bgh[tt], 0.f));
                    const float h1 = dh * sH[rb + v][col];
                    hn[tt][v] = h1;
                    sA[rb + v][kD + col] = (__bf16)h1;   // comb h-slice
                }
            }
        }
        __syncthreads();

        // ---- phase 2: z and r gates: 4 independent WMMA chains share each A fragment ----
        {
            v8f az[2] = {}, ar[2] = {};
#pragma unroll
            for (int kc = 0; kc < 24; ++kc) {
                v16bf a = load_a_lds(&sA[0][0], kAPAD, kc * 32, lane);
                az[0] = WMMA_BF16(a, load_b_glb(WzB, kK, c0,      kc * 32, lane), az[0]);
                az[1] = WMMA_BF16(a, load_b_glb(WzB, kK, c0 + 16, kc * 32, lane), az[1]);
                ar[0] = WMMA_BF16(a, load_b_glb(WrB, kK, c0,      kc * 32, lane), ar[0]);
                ar[1] = WMMA_BF16(a, load_b_glb(WrB, kK, c0 + 16, kc * 32, lane), ar[1]);
            }
#pragma unroll
            for (int tt = 0; tt < 2; ++tt) {
                const int col = c0 + tt * 16 + (lane & 15);
                const int rb  = (lane >> 4) << 3;
#pragma unroll
                for (int v = 0; v < 8; ++v) {
                    const float z = 1.f / (1.f + __expf(-(az[tt][v] + c_bz[tt])));
                    const float r = 1.f / (1.f + __expf(-(ar[tt][v] + c_br[tt])));
                    zz[tt][v] = z;
                    sM2[rb + v][col] = (__bf16)(r * hn[tt][v]);
                }
            }
        }
        __syncthreads();

        // ---- phase 3: h_tilde = tanh(comb_r @ Wh.T + bh); gated state update ----
        {
            v8f ah[2] = {};
#pragma unroll
            for (int kc = 0; kc < 24; ++kc) {
                v16bf a = (kc >= 8 && kc < 16)
                            ? load_a_lds(&sM2[0][0], kDPAD, (kc - 8) * 32, lane)
                            : load_a_lds(&sA[0][0],  kAPAD, kc * 32, lane);
                ah[0] = WMMA_BF16(a, load_b_glb(WhB, kK, c0,      kc * 32, lane), ah[0]);
                ah[1] = WMMA_BF16(a, load_b_glb(WhB, kK, c0 + 16, kc * 32, lane), ah[1]);
            }
#pragma unroll
            for (int tt = 0; tt < 2; ++tt) {
                const int col = c0 + tt * 16 + (lane & 15);
                const int rb  = (lane >> 4) << 3;
#pragma unroll
                for (int v = 0; v < 8; ++v) {
                    const float ht = tanhf(ah[tt][v] + c_bh[tt]);
                    const float z  = zz[tt][v];
                    sH[rb + v][col] = (1.f - z) * hn[tt][v] + z * ht;
                }
            }
        }
        __syncthreads();
    }

    // ---- classifier + softmax: 16 rows x 2 classes, one thread per row ----
    if (tid < kMT) {
        float l0 = bfc[0], l1 = bfc[1];
        for (int k = 0; k < kH; ++k) {
            const float hv = sH[tid][k];
            l0 += hv * Wfc[k];
            l1 += hv * Wfc[kH + k];
        }
        const float mx = fmaxf(l0, l1);
        const float e0 = __expf(l0 - mx), e1 = __expf(l1 - mx);
        const float s  = e0 + e1;
        out[(b0 + tid) * 2 + 0] = e0 / s;
        out[(b0 + tid) * 2 + 1] = e1 / s;
    }
}

extern "C" void kernel_launch(void* const* d_in, const int* in_sizes, int n_in,
                              void* d_out, int out_size, void* d_ws, size_t ws_size,
                              hipStream_t stream) {
    const float* inp   = (const float*)d_in[0];
    const float* Xmean = (const float*)d_in[1];
    const float* Wg_x  = (const float*)d_in[2];
    const float* bg_x  = (const float*)d_in[3];
    const float* Wg_h  = (const float*)d_in[4];
    const float* bg_h  = (const float*)d_in[5];
    const float* Wz    = (const float*)d_in[6];
    const float* bz    = (const float*)d_in[7];
    const float* Wr    = (const float*)d_in[8];
    const float* br    = (const float*)d_in[9];
    const float* Wh    = (const float*)d_in[10];
    const float* bh    = (const float*)d_in[11];
    const float* Wfc   = (const float*)d_in[12];
    const float* bfc   = (const float*)d_in[13];
    float*       out   = (float*)d_out;
    __bf16*      wsb   = (__bf16*)d_ws;

    grud_prep<<<256, 256, 0, stream>>>(Wg_h, Wz, Wr, Wh, wsb);
    grud_main<<<kB / kMT, 256, kDynLDS, stream>>>(inp, Xmean, Wg_x, bg_x, bg_h,
                                                  bz, br, bh, Wfc, bfc, wsb, out);
}
